// RWNNLM_17497696764546
// MI455X (gfx1250) — matrix operand order repaired and checked by reading, not compiled, not run on hardware
//
#include <hip/hip_runtime.h>
#include <hip/hip_bf16.h>
#include <math.h>

typedef __attribute__((ext_vector_type(16))) _Float16 v16h;
typedef __attribute__((ext_vector_type(8)))  float    v8f;
typedef __attribute__((ext_vector_type(4)))  float    f4;
typedef __attribute__((ext_vector_type(4)))  unsigned int u32x4;
typedef __attribute__((ext_vector_type(4)))  int      i32x4;
typedef __attribute__((ext_vector_type(8)))  int      i32x8;

#define BB      2048
#define VV      50257
#define TT      64
#define DD      32
#define N_IN    128
#define N_OUT   128
#define N_HID   1500
#define HIDPL   300
#define KK      45
#define N_BIAS  2
#define VALS_ROWS (N_IN + N_BIAS + N_HID)   // 1630

// ---------------- WMMA fragment helpers (wave32, 16x16x32 f16) ----------------
// A (16x32, MxK): lane -> M = lane&15 ; K = (v<4 ? 2v : 16+2(v-4)) + 8*(lane>=16) + (e&1)
__device__ __forceinline__ int a_k_map(int e, int laneHi8) {
    int v = e >> 1;
    return (v < 4 ? 2 * v : 16 + 2 * (v - 4)) + laneHi8 + (e & 1);
}
// B (32x16, KxN): lane -> N = lane&15 ; K = same map as A (a_k_map)

// =========================================================================
// Kernel 1: x[b,n] = sum_td gather(token_emb, ids)[b,td] * in_w[n,td] + in_b[n]
// =========================================================================
__global__ __launch_bounds__(256)
void k_in_gemm(const int* __restrict__ ids, const float* __restrict__ emb,
               const float* __restrict__ in_w, const float* __restrict__ in_b,
               float* __restrict__ x_out) {
    __shared__ _Float16 sA[64 * 32];
    __shared__ _Float16 sB[128 * 32];
    const int tid  = threadIdx.x;
    const int lane = tid & 31, wave = tid >> 5;
    const int rowBase = blockIdx.x * 64;
    const int rBlk = wave >> 1;
    const int cGrp = wave & 1;
    const int m    = lane & 15;
    const int hi8  = (lane & 16) ? 8 : 0;
    const int hi16 = (lane & 16) ? 16 : 0;

    v8f acc[4];
    #pragma unroll
    for (int i = 0; i < 4; ++i)
        #pragma unroll
        for (int r = 0; r < 8; ++r) acc[i][r] = 0.0f;

    for (int t = 0; t < TT; ++t) {
        {
            const int r  = tid >> 2;
            const int d0 = (tid & 3) * 8;
            const int id = ids[(rowBase + r) * TT + t];
            const float* src = emb + (size_t)id * DD + d0;
            #pragma unroll
            for (int j = 0; j < 8; ++j) sA[r * 32 + d0 + j] = (_Float16)src[j];
        }
        {
            const int n  = tid >> 1;
            const int d0 = (tid & 1) * 16;
            const float* src = in_w + (size_t)n * (TT * DD) + t * DD + d0;
            #pragma unroll
            for (int j = 0; j < 16; ++j) sB[n * 32 + d0 + j] = (_Float16)src[j];
        }
        __syncthreads();

        v16h a;
        #pragma unroll
        for (int e = 0; e < 16; ++e) a[e] = sA[(rBlk * 16 + m) * 32 + a_k_map(e, hi8)];
        #pragma unroll
        for (int nt = 0; nt < 4; ++nt) {
            const int n = cGrp * 64 + nt * 16 + m;
            v16h b;
            #pragma unroll
            for (int e = 0; e < 16; ++e) b[e] = sB[n * 32 + a_k_map(e, hi8)];
            acc[nt] = __builtin_amdgcn_wmma_f32_16x16x32_f16(
                false, a, false, b, (short)0, acc[nt], false, false);
        }
        __syncthreads();
    }

    #pragma unroll
    for (int nt = 0; nt < 4; ++nt) {
        const int col  = cGrp * 64 + nt * 16 + m;
        const float bv = in_b[col];
        #pragma unroll
        for (int r = 0; r < 8; ++r) {
            const int row = rowBase + rBlk * 16 + r + hi8;
            x_out[(size_t)row * N_IN + col] = acc[nt][r] + bv;
        }
    }
}

// =========================================================================
// Kernel 2: random-wired DAG forward. 8 samples/WG; vals LDS-resident.
// =========================================================================
__global__ __launch_bounds__(256)
void k_rwnn(const float* __restrict__ x, const int* __restrict__ src_idx,
            const float* __restrict__ rwnn_w, float* __restrict__ h_out) {
    __shared__ float vals[VALS_ROWS * 8];
    const int tid = threadIdx.x;
    const int s   = tid & 7;
    const int ng  = tid >> 3;
    const int bBase = blockIdx.x * 8;

    for (int n = ng; n < N_IN + N_BIAS; n += 32)
        vals[n * 8 + s] = (n < N_IN) ? x[(size_t)(bBase + s) * N_IN + n] : 1.0f;
    __syncthreads();

    int off = 0;
    for (int layer = 0; layer < 5; ++layer) {
        for (int n = ng; n < HIDPL; n += 32) {
            const int*   sr = src_idx + (size_t)(off + n) * KK;
            const float* wr = rwnn_w  + (size_t)(off + n) * KK;
            float acc = 0.0f;
            #pragma unroll 5
            for (int k = 0; k < KK; ++k) acc += vals[sr[k] * 8 + s] * wr[k];
            vals[(N_IN + N_BIAS + off + n) * 8 + s] = tanhf(acc);
        }
        __syncthreads();
        off += HIDPL;
    }
    for (int n = ng; n < N_OUT; n += 32) {
        const int*   sr = src_idx + (size_t)(N_HID + n) * KK;
        const float* wr = rwnn_w  + (size_t)(N_HID + n) * KK;
        float acc = 0.0f;
        #pragma unroll 5
        for (int k = 0; k < KK; ++k) acc += vals[sr[k] * 8 + s] * wr[k];
        h_out[(size_t)(bBase + s) * N_OUT + n] = tanhf(acc);
    }
}

// =========================================================================
// Kernel 3: logits[b,v] = sum_n h[b,n]*out_w[v,n] + out_b[v]
// 64(B) x 64(V) C-tile per WG. out_w tile DMA'd by the TDM into LDS (f32),
// h tile staged as f16. C restaged through LDS -> coalesced NT b128 stores.
// =========================================================================
#define CT 64
__global__ __launch_bounds__(256)
void k_out_gemm(const float* __restrict__ h, const float* __restrict__ out_w,
                const float* __restrict__ out_b, float* __restrict__ out) {
    __shared__ _Float16 sH[64 * 128];    // 16 KB
    __shared__ float    sWf[CT * 128];   // 32 KB, later reused as 64x64 f32 C tile
    const int tid = threadIdx.x;
    const int lane = tid & 31, wave = tid >> 5;
    const int rowBase = blockIdx.y * 64;
    const int colBase = blockIdx.x * CT;

    // ---- stage h tile (f32 -> f16) ----
    for (int i = tid; i < 64 * 128; i += 256)
        sH[i] = (_Float16)h[(size_t)(rowBase + (i >> 7)) * N_OUT + (i & 127)];

    // ---- TDM: DMA out_w tile [CT rows x 128 cols] f32 into sWf ----
    if (wave == 0) {
        const unsigned long long gaddr =
            (unsigned long long)(uintptr_t)(out_w + (size_t)colBase * N_OUT);
        const unsigned int ldsa = (unsigned int)(unsigned long long)(uintptr_t)&sWf[0];
        const unsigned int rem  = (unsigned int)(VV - colBase);   // rows left in tensor

        u32x4 g0;
        g0[0] = 1u;                                   // count=1, user descriptor
        g0[1] = ldsa;                                 // lds_addr
        g0[2] = (unsigned int)(gaddr & 0xFFFFFFFFull);            // global_addr[31:0]
        g0[3] = (unsigned int)((gaddr >> 32) & 0x01FFFFFFull)     // global_addr[56:32]
              | 0x80000000u;                                      // type=2 ("image")
        i32x8 g1;
        g1[0] = (int)(2u << 16);                      // data_size=4B, wg_mask=0
        g1[1] = (int)(128u << 16);                    // tensor_dim0[15:0]=128
        g1[2] = (int)((rem & 0xFFFFu) << 16);         // tensor_dim0 hi=0 | tensor_dim1 lo
        g1[3] = (int)(128u << 16);                    // tensor_dim1 hi=0 | tile_dim0=128
        g1[4] = (int)CT;                              // tile_dim1=64, tile_dim2=0
        g1[5] = 128;                                  // tensor_dim0_stride lo = 128
        g1[6] = 0;                                    // stride hi / dim1_stride lo
        g1[7] = 0;                                    // dim1_stride hi
        i32x4 gz = {0, 0, 0, 0};                      // groups 2/3 unused (2-D tensor)
#if defined(__clang_major__) && (__clang_major__ >= 23)
        i32x8 gz8 = {0, 0, 0, 0, 0, 0, 0, 0};
        __builtin_amdgcn_tensor_load_to_lds(g0, g1, gz, gz, gz8, 0);
#else
        __builtin_amdgcn_tensor_load_to_lds(g0, g1, gz, gz, 0);
#endif
        __builtin_amdgcn_s_wait_tensorcnt(0);
    }
    __syncthreads();

    // ---- WMMA: each wave computes 16 rows x 32 cols (2 accumulators) ----
    const int rBlk  = wave >> 1;         // 0..3
    const int cPair = wave & 1;          // 0..1
    const int m     = lane & 15;
    const int hi8   = (lane & 16) ? 8 : 0;
    const int hi16  = (lane & 16) ? 16 : 0;

    v8f acc[2];
    #pragma unroll
    for (int i = 0; i < 2; ++i)
        #pragma unroll
        for (int r = 0; r < 8; ++r) acc[i][r] = 0.0f;

    #pragma unroll
    for (int kb = 0; kb < 128; kb += 32) {
        v16h a;
        #pragma unroll
        for (int e = 0; e < 16; ++e)
            a[e] = sH[(rBlk * 16 + m) * 128 + kb + a_k_map(e, hi8)];
        #pragma unroll
        for (int nt = 0; nt < 2; ++nt) {
            const int n = cPair * 32 + nt * 16 + m;   // V index within tile
            v16h b;
            #pragma unroll
            for (int e = 0; e < 16; ++e)
                b[e] = (_Float16)sWf[n * 128 + kb + a_k_map(e, hi8)];
            acc[nt] = __builtin_amdgcn_wmma_f32_16x16x32_f16(
                false, a, false, b, (short)0, acc[nt], false, false);
        }
    }
    __syncthreads();                      // everyone done reading sWf

    // ---- restage C through LDS (reuse sWf) ----
    float* sC = sWf;                      // [64 rows][64 cols]
    #pragma unroll
    for (int nt = 0; nt < 2; ++nt) {
        const int c   = cPair * 32 + nt * 16 + m;
        const int col = colBase + c;
        const float bv = (col < VV) ? out_b[col] : 0.0f;
        #pragma unroll
        for (int r = 0; r < 8; ++r)
            sC[(rBlk * 16 + r + hi8) * CT + c] = acc[nt][r] + bv;
    }
    __syncthreads();

    // ---- coalesced non-temporal stores ----
    if (colBase + CT <= VV) {             // full tile: 128-bit NT stores
        for (int i = tid; i < 64 * CT / 4; i += 256) {
            const int row = i >> 4;
            const int c4  = (i & 15) << 2;
            f4 val = *(const f4*)&sC[row * CT + (i & 15) * 4];
            (void)c4;
            __builtin_nontemporal_store(
                val, (f4*)(out + (size_t)(rowBase + row) * VV + colBase + (i & 15) * 4));
        }
    } else {                              // vocab tail: guarded scalar stores
        for (int i = tid; i < 64 * CT; i += 256) {
            const int row = i >> 6;
            const int col = colBase + (i & 63);
            if (col < VV)
                __builtin_nontemporal_store(
                    sC[i], out + (size_t)(rowBase + row) * VV + col);
        }
    }
}

// =========================================================================
extern "C" void kernel_launch(void* const* d_in, const int* in_sizes, int n_in,
                              void* d_out, int out_size, void* d_ws, size_t ws_size,
                              hipStream_t stream) {
    const int*   ids       = (const int*)  d_in[0];
    const float* token_emb = (const float*)d_in[1];
    const float* in_w      = (const float*)d_in[2];
    const float* in_b      = (const float*)d_in[3];
    const int*   src_idx   = (const int*)  d_in[4];
    const float* rwnn_w    = (const float*)d_in[5];
    const float* out_w     = (const float*)d_in[6];
    const float* out_b     = (const float*)d_in[7];
    float* outp = (float*)d_out;

    float* x = (float*)d_ws;                                   // [B, 128]
    float* h = (float*)((char*)d_ws + (size_t)BB * N_IN * 4);  // [B, 128]

    k_in_gemm<<<dim3(BB / 64), dim3(256), 0, stream>>>(ids, token_emb, in_w, in_b, x);
    k_rwnn<<<dim3(BB / 8), dim3(256), 0, stream>>>(x, src_idx, rwnn_w, h);
    k_out_gemm<<<dim3((VV + CT - 1) / CT, BB / 64), dim3(256), 0, stream>>>(h, out_w, out_b, outp);
}
